// GroupMamba_55078660603917
// MI455X (gfx1250) — matrix operand order repaired and checked by reading, not compiled, not run on hardware
//
#include <hip/hip_runtime.h>
#include <hip/hip_bf16.h>
#include <math.h>

// ---------- types for CDNA5 WMMA / TDM ----------
typedef _Float16 v8h  __attribute__((ext_vector_type(8)));
typedef _Float16 v16h __attribute__((ext_vector_type(16)));
typedef float    v8f  __attribute__((ext_vector_type(8)));
typedef unsigned u32x4 __attribute__((ext_vector_type(4)));
typedef int      i32x4 __attribute__((ext_vector_type(4)));
typedef int      i32x8 __attribute__((ext_vector_type(8)));

union F16Frag { v16h v; v8h h2[2]; };

// Tensor Data Mover availability (device pass); fall back to manual staging.
#if defined(__has_builtin)
#  if __has_builtin(__builtin_amdgcn_tensor_load_to_lds)
#    define HAVE_TDM 1
#  else
#    define HAVE_TDM 0
#  endif
#else
#  define HAVE_TDM 0
#endif
#if defined(__has_include)
#  if __has_include(<hip/amd_detail/amd_gfx1250_TDM.h>)
#    define TDM_6ARG 1
#  else
#    define TDM_6ARG 0
#  endif
#else
#  define TDM_6ARG 0
#endif

#define LN_EPS 1e-5f

// problem dims (fixed by reference setup)
#define BB   16
#define CC   512
#define HH   48
#define WW2  48
#define LL   (HH * WW2)     // 2304
#define DD   128            // d = C/4
#define MM   (BB * LL)      // 36864 rows

__device__ __forceinline__ float siluf(float x) { return x / (1.0f + expf(-x)); }

__device__ __forceinline__ float blockReduceSum(float v, int tid, int nthr) {
    __shared__ float red[256];
    red[tid] = v;
    __syncthreads();
    for (int s = nthr >> 1; s > 0; s >>= 1) {
        if (tid < s) red[tid] += red[tid + s];
        __syncthreads();
    }
    float r = red[0];
    __syncthreads();
    return r;
}

// ---------- 1) transpose x (B,C,L) -> xs (B,L,C) ----------
__global__ void k_transpose(const float* __restrict__ x, float* __restrict__ xs) {
    __shared__ float tile[32][33];
    int b  = blockIdx.z;
    int l0 = blockIdx.x * 32;
    int c0 = blockIdx.y * 32;
    const float* xin = x + (size_t)b * CC * LL;
    float*       out = xs + (size_t)b * LL * CC;
    for (int i = threadIdx.y; i < 32; i += 8)
        tile[i][threadIdx.x] = xin[(size_t)(c0 + i) * LL + l0 + threadIdx.x];
    __syncthreads();
    for (int i = threadIdx.y; i < 32; i += 8)
        out[(size_t)(l0 + i) * CC + c0 + threadIdx.x] = tile[threadIdx.x][i];
}

// ---------- 2) row layernorm over C (in place) + f16 copy ----------
__global__ void k_ln_rows(float* __restrict__ xs, _Float16* __restrict__ xs16,
                          const float* __restrict__ w, const float* __restrict__ bsh) {
    int tid = threadIdx.x;
    size_t base = (size_t)blockIdx.x * CC;
    float v[4], s = 0.f;
#pragma unroll
    for (int j = 0; j < 4; ++j) { v[j] = xs[base + tid + j * 128]; s += v[j]; }
    float mean = blockReduceSum(s, tid, 128) * (1.0f / CC);
    s = 0.f;
#pragma unroll
    for (int j = 0; j < 4; ++j) { float d = v[j] - mean; s += d * d; }
    float var = blockReduceSum(s, tid, 128) * (1.0f / CC);
    float inv = rsqrtf(var + LN_EPS);
#pragma unroll
    for (int j = 0; j < 4; ++j) {
        int c = tid + j * 128;
        float o = (v[j] - mean) * inv * w[c] + bsh[c];
        xs[base + c]   = o;
        xs16[base + c] = (_Float16)o;
    }
}

// ---------- 3) z = mean over L of xs ----------
__global__ void k_zmean(const float* __restrict__ xs, float* __restrict__ z) {
    int i = blockIdx.x * 256 + threadIdx.x;          // i in [0, B*C)
    int b = i >> 9, c = i & 511;
    float acc = 0.f;
    for (int l = 0; l < LL; ++l)
        acc += xs[((size_t)b * LL + l) * CC + c];
    z[i] = acc * (1.0f / LL);
}

// ---------- 4) squeeze-excite ----------
__global__ void k_se(const float* __restrict__ z,
                     const float* __restrict__ w1, const float* __restrict__ b1,
                     const float* __restrict__ w2, const float* __restrict__ b2,
                     float* __restrict__ se) {
    __shared__ float hid[32];
    int b = blockIdx.x, tid = threadIdx.x;
    if (tid < 32) {
        float s = b1[tid];
        for (int c = 0; c < CC; ++c) s += z[b * CC + c] * w1[c * 32 + tid];
        hid[tid] = fmaxf(s, 0.f);
    }
    __syncthreads();
    for (int c = tid; c < CC; c += 128) {
        float s = b2[c];
#pragma unroll
        for (int j = 0; j < 32; ++j) s += hid[j] * w2[j * CC + c];
        se[b * CC + c] = 1.0f / (1.0f + expf(-s));
    }
}

// ---------- 5) fp32 -> f16 weight convert ----------
__global__ void k_f32_to_f16(const float* __restrict__ src, _Float16* __restrict__ dst, int n) {
    int i = blockIdx.x * 256 + threadIdx.x;
    if (i < n) dst[i] = (_Float16)src[i];
}

// ---------- 6) WMMA GEMM: C[M x N] = A[M x K](f16) * Bw[K x N](f16) (+bias) ----------
// block = 128 thr (4 waves), tile 128x64, BK = 32; wave w -> rows [32w,32w+32)
// A tile staged via Tensor Data Mover (wave 0 issues, pad feature recreates the
// 40-half padded LDS rows); B tile staged transposed by all threads.
__global__ __launch_bounds__(128)
void k_gemm_wmma(const _Float16* __restrict__ A, int lda,
                 const _Float16* __restrict__ Bw, int ldb,
                 float* __restrict__ Co, int ldc,
                 const float* __restrict__ bias, int K,
                 int outTransposed, int Lseq, int Nfull) {
    __shared__ __align__(16) _Float16 As[128][40];
    __shared__ __align__(16) _Float16 Bt[64][40];
    int tid  = threadIdx.x;
    int wv   = tid >> 5, lane = tid & 31;
    int hi   = lane >> 4, l16 = lane & 15;
    int mBase = blockIdx.y * 128, nBase = blockIdx.x * 64;

    v8f acc[2][4];
#pragma unroll
    for (int f = 0; f < 2; ++f)
#pragma unroll
        for (int t = 0; t < 4; ++t)
#pragma unroll
            for (int e = 0; e < 8; ++e) acc[f][t][e] = 0.f;

    int akb = hi * 8;
    int bkb = hi * 8;
#if HAVE_TDM
    unsigned ldsA = (unsigned)(unsigned long long)(void*)&As[0][0];
#endif

    for (int k0 = 0; k0 < K; k0 += 32) {
#if HAVE_TDM
        if (wv == 0) {
            // D# group 0: count=1 | lds_addr | global_addr(57b) | type=2
            unsigned long long ga = (unsigned long long)(A + (size_t)mBase * lda + k0);
            u32x4 g0;
            g0[0] = 1u;
            g0[1] = ldsA;
            g0[2] = (unsigned)ga;
            g0[3] = (unsigned)((ga >> 32) & 0x01FFFFFFull) | 0x80000000u;
            // D# group 1: data_size=2B, pad_enable, pad_interval=16 DW, pad_amount=4 DW,
            // tensor_dim0=lda, tensor_dim1=large, tile 32x128, dim0 stride=lda
            unsigned td0 = (unsigned)lda, td1 = 1u << 20;
            i32x8 g1;
            g1[0] = (int)((1u << 16) | (1u << 20) | (3u << 22) | (3u << 25));
            g1[1] = (int)((td0 & 0xFFFFu) << 16);
            g1[2] = (int)((td0 >> 16) | ((td1 & 0xFFFFu) << 16));
            g1[3] = (int)((td1 >> 16) | (32u << 16));
            g1[4] = (int)128u;                  // tile_dim1=128, tile_dim2=0
            g1[5] = (int)td0;                   // tensor_dim0_stride lo
            g1[6] = 0; g1[7] = 0;
            i32x4 z4 = {0, 0, 0, 0};
#if TDM_6ARG
            i32x8 z8 = {0, 0, 0, 0, 0, 0, 0, 0};
            __builtin_amdgcn_tensor_load_to_lds(g0, g1, z4, z4, z8, 0);
#else
            __builtin_amdgcn_tensor_load_to_lds(g0, g1, z4, z4, 0);
#endif
        }
#else
        // manual A staging: 128x32 halves, four 16B chunks per thread
#pragma unroll
        for (int i = 0; i < 4; ++i) {
            int c   = tid + i * 128;
            int row = c >> 2, cc = c & 3;
            *(v8h*)&As[row][cc * 8] =
                *(const v8h*)(A + (size_t)(mBase + row) * lda + k0 + cc * 8);
        }
#endif
        // stage B tile transposed: Bt[n][k]
        for (int i = tid; i < 2048; i += 128) {
            int kk = i >> 6, n = i & 63;
            Bt[n][kk] = Bw[(size_t)(k0 + kk) * ldb + nBase + n];
        }
        if (k0 + 32 < K)
            __builtin_prefetch(Bw + (size_t)(k0 + 32 + (tid >> 6)) * ldb + nBase, 0, 1);
#if HAVE_TDM
        if (wv == 0) {
#if __has_builtin(__builtin_amdgcn_s_wait_tensorcnt)
            __builtin_amdgcn_s_wait_tensorcnt(0);
#else
            asm volatile("s_wait_tensorcnt 0x0" ::: "memory");
#endif
        }
#endif
        __syncthreads();

        F16Frag a0, a1, bf;
        a0.h2[0] = *(const v8h*)&As[wv * 32 + l16][akb];
        a0.h2[1] = *(const v8h*)&As[wv * 32 + l16][akb + 16];
        a1.h2[0] = *(const v8h*)&As[wv * 32 + 16 + l16][akb];
        a1.h2[1] = *(const v8h*)&As[wv * 32 + 16 + l16][akb + 16];
#pragma unroll
        for (int t = 0; t < 4; ++t) {
            int bcol = t * 16 + l16;
            bf.h2[0] = *(const v8h*)&Bt[bcol][bkb];
            bf.h2[1] = *(const v8h*)&Bt[bcol][bkb + 16];
            acc[0][t] = __builtin_amdgcn_wmma_f32_16x16x32_f16(
                false, a0.v, false, bf.v, (short)0, acc[0][t], false, false);
            acc[1][t] = __builtin_amdgcn_wmma_f32_16x16x32_f16(
                false, a1.v, false, bf.v, (short)0, acc[1][t], false, false);
        }
        __syncthreads();
    }

#pragma unroll
    for (int f = 0; f < 2; ++f) {
#pragma unroll
        for (int t = 0; t < 4; ++t) {
            int col = nBase + t * 16 + l16;
            float bv = bias ? bias[col] : 0.f;
#pragma unroll
            for (int r = 0; r < 8; ++r) {
                int row = mBase + wv * 32 + f * 16 + hi * 8 + r;
                float val = acc[f][t][r] + bv;
                if (!outTransposed) {
                    Co[(size_t)row * ldc + col] = val;
                } else {
                    int b = row / Lseq, l = row % Lseq;
                    Co[((size_t)b * Nfull + col) * Lseq + l] = val;
                }
            }
        }
    }
}

// ---------- 7) depthwise 3x3 conv + bias + silu ----------
__global__ void k_dwconv_silu(const float* __restrict__ xz,
                              const float* __restrict__ cw, const float* __restrict__ cb,
                              float* __restrict__ xc) {
    int pos = blockIdx.x, b = blockIdx.y, ch = threadIdx.x;
    int h = pos / WW2, w = pos % WW2;
    float s = cb[ch];
#pragma unroll
    for (int ky = 0; ky < 3; ++ky) {
#pragma unroll
        for (int kx = 0; kx < 3; ++kx) {
            int hh = h + ky - 1, ww = w + kx - 1;
            if (hh >= 0 && hh < HH && ww >= 0 && ww < WW2)
                s += xz[((size_t)b * LL + hh * WW2 + ww) * 256 + ch] * cw[ch * 9 + ky * 3 + kx];
        }
    }
    xc[((size_t)b * LL + pos) * DD + ch] = siluf(s);
}

// ---------- 8) x_dbl: xd[b,pos,r] = sum_ch Wx[r,ch]*u ----------
__global__ void k_xdbl(const float* __restrict__ xc, const float* __restrict__ Wx,
                       float* __restrict__ xd) {
    __shared__ float sv[128];
    int tid = threadIdx.x;
    sv[tid] = xc[(size_t)blockIdx.x * DD + tid];
    __syncthreads();
    for (int r = 0; r < 10; ++r) {
        float tot = blockReduceSum(sv[tid] * Wx[r * DD + tid], tid, 128);
        if (tid == 0) xd[(size_t)blockIdx.x * 10 + r] = tot;
    }
}

// ---------- 9) selective scan over L (channel-parallel, direction-remapped) ----------
__global__ void k_scan(const float* __restrict__ xc, const float* __restrict__ xd,
                       const float* __restrict__ Wdt, const float* __restrict__ dtb,
                       const float* __restrict__ A_log, const float* __restrict__ Dpv,
                       float* __restrict__ y, int k) {
    int b = blockIdx.x, ch = threadIdx.x;
    float wdt[8];
#pragma unroll
    for (int r = 0; r < 8; ++r) wdt[r] = Wdt[ch * 8 + r];
    float dbias = dtb[ch];
    float A  = -expf(A_log[ch]);
    float Dd = Dpv[ch];
    __shared__ float sd[10];
    float h = 0.f;
    for (int t = 0; t < LL; ++t) {
        int i   = (k & 1) ? (LL - 1 - t) : t;
        int pos = (k >= 2) ? ((i % HH) * WW2 + (i / HH)) : i;
        if (ch < 10) sd[ch] = xd[((size_t)b * LL + pos) * 10 + ch];
        __syncthreads();
        float xr = dbias;
#pragma unroll
        for (int r = 0; r < 8; ++r) xr += sd[r] * wdt[r];
        float delta = (xr > 20.f) ? xr : log1pf(expf(xr));
        float u  = xc[((size_t)b * LL + pos) * DD + ch];
        float dA = expf(delta * A);
        h = dA * h + delta * u * sd[8];
        y[((size_t)b * LL + pos) * DD + ch] = h * sd[9] + Dd * u;
        __syncthreads();
    }
}

// ---------- 10) out layernorm over d * silu(z-branch) -> f16 ----------
__global__ void k_outln(const float* __restrict__ y, const float* __restrict__ xz,
                        const float* __restrict__ nw, const float* __restrict__ nb,
                        _Float16* __restrict__ g16) {
    int tid = threadIdx.x;
    size_t row = blockIdx.x;
    float v = y[row * DD + tid];
    float mean = blockReduceSum(v, tid, 128) * (1.0f / DD);
    float d = v - mean;
    float var = blockReduceSum(d * d, tid, 128) * (1.0f / DD);
    float g = d * rsqrtf(var + LN_EPS) * nw[tid] + nb[tid];
    float zb = xz[row * 256 + 128 + tid];
    g *= siluf(zb);
    g16[row * DD + tid] = (_Float16)g;
}

// ---------- 11) xm * skip * xh * se, layernorm over C -> f16 ----------
__global__ void k_mixln(const float* __restrict__ xm, const float* __restrict__ xh,
                        const float* __restrict__ se, const float* __restrict__ ss,
                        const float* __restrict__ w, const float* __restrict__ bsh,
                        _Float16* __restrict__ xmn) {
    int tid = threadIdx.x;
    size_t row = blockIdx.x;
    int b = (int)(row / LL);
    float sc = ss[0];
    float v[4], s = 0.f;
#pragma unroll
    for (int j = 0; j < 4; ++j) {
        int c = tid + j * 128;
        v[j] = xm[row * CC + c] * sc * xh[row * CC + c] * se[b * CC + c];
        s += v[j];
    }
    float mean = blockReduceSum(s, tid, 128) * (1.0f / CC);
    s = 0.f;
#pragma unroll
    for (int j = 0; j < 4; ++j) { float d = v[j] - mean; s += d * d; }
    float var = blockReduceSum(s, tid, 128) * (1.0f / CC);
    float inv = rsqrtf(var + LN_EPS);
#pragma unroll
    for (int j = 0; j < 4; ++j) {
        int c = tid + j * 128;
        xmn[row * CC + c] = (_Float16)((v[j] - mean) * inv * w[c] + bsh[c]);
    }
}

// ============================== host launcher ==============================
extern "C" void kernel_launch(void* const* d_in, const int* in_sizes, int n_in,
                              void* d_out, int out_size, void* d_ws, size_t ws_size,
                              hipStream_t stream) {
    const float* x       = (const float*)d_in[0];
    const float* norm_w  = (const float*)d_in[1];
    const float* norm_b  = (const float*)d_in[2];
    const float* fc1_w   = (const float*)d_in[3];
    const float* fc1_b   = (const float*)d_in[4];
    const float* fc2_w   = (const float*)d_in[5];
    const float* fc2_b   = (const float*)d_in[6];
    const float* Win     = (const float*)d_in[7];
    const float* conv_w  = (const float*)d_in[8];
    const float* conv_b  = (const float*)d_in[9];
    const float* Wx      = (const float*)d_in[10];
    const float* Wdt     = (const float*)d_in[11];
    const float* dt_bias = (const float*)d_in[12];
    const float* A_log   = (const float*)d_in[13];
    const float* Dp      = (const float*)d_in[14];
    const float* onorm_w = (const float*)d_in[15];
    const float* onorm_b = (const float*)d_in[16];
    const float* Wout    = (const float*)d_in[17];
    const float* proj_w  = (const float*)d_in[18];
    const float* proj_b  = (const float*)d_in[19];
    const float* skip_sc = (const float*)d_in[20];
    float* out = (float*)d_out;

    // carve workspace (all sizes already 256B multiples)
    char* ws = (char*)d_ws;
    size_t off = 0;
    float*     xsln  = (float*)(ws + off);      off += (size_t)MM * CC * 4;   // LN'ed xs (= xh)
    _Float16*  xs16  = (_Float16*)(ws + off);   off += (size_t)MM * CC * 2;
    float*     xz    = (float*)(ws + off);      off += (size_t)MM * 256 * 4;  // per-k
    float*     xc    = (float*)(ws + off);      off += (size_t)MM * DD * 4;   // per-k
    float*     xd    = (float*)(ws + off);      off += (size_t)MM * 10 * 4;   // per-k
    float*     yb    = (float*)(ws + off);      off += (size_t)MM * DD * 4;   // per-k
    _Float16*  g16   = (_Float16*)(ws + off);   off += (size_t)MM * DD * 2;   // per-k
    float*     xm    = (float*)(ws + off);      off += (size_t)MM * CC * 4;
    _Float16*  xmn   = (_Float16*)(ws + off);   off += (size_t)MM * CC * 2;
    float*     zbuf  = (float*)(ws + off);      off += (size_t)BB * CC * 4;
    float*     sebuf = (float*)(ws + off);      off += (size_t)BB * CC * 4;
    _Float16*  win16 = (_Float16*)(ws + off);   off += (size_t)4 * DD * 256 * 2;
    _Float16*  wout16= (_Float16*)(ws + off);   off += (size_t)4 * DD * DD * 2;
    _Float16*  proj16= (_Float16*)(ws + off);   off += (size_t)CC * CC * 2;
    (void)off; (void)ws_size; (void)in_sizes; (void)n_in; (void)out_size;

    // 1) transpose + 2) input LN + f16
    k_transpose<<<dim3(LL / 32, CC / 32, BB), dim3(32, 8), 0, stream>>>(x, xsln);
    k_ln_rows<<<MM, 128, 0, stream>>>(xsln, xs16, norm_w, norm_b);

    // 3) channel means, 4) SE
    k_zmean<<<(BB * CC) / 256, 256, 0, stream>>>(xsln, zbuf);
    k_se<<<BB, 128, 0, stream>>>(zbuf, fc1_w, fc1_b, fc2_w, fc2_b, sebuf);

    // 5) weight conversions to f16
    k_f32_to_f16<<<(4 * DD * 256 + 255) / 256, 256, 0, stream>>>(Win, win16, 4 * DD * 256);
    k_f32_to_f16<<<(4 * DD * DD + 255) / 256, 256, 0, stream>>>(Wout, wout16, 4 * DD * DD);
    k_f32_to_f16<<<(CC * CC + 255) / 256, 256, 0, stream>>>(proj_w, proj16, CC * CC);

    // 6) per-direction SS2D
    for (int k = 0; k < 4; ++k) {
        // xz = xs[:, :, k*d:(k+1)*d] @ Win[k]   (M x 128) @ (128 x 256)
        k_gemm_wmma<<<dim3(256 / 64, MM / 128), 128, 0, stream>>>(
            xs16 + k * DD, CC, win16 + (size_t)k * DD * 256, 256,
            xz, 256, nullptr, DD, 0, LL, 256);
        // depthwise conv + silu -> u (spatial row-major)
        k_dwconv_silu<<<dim3(LL, BB), 128, 0, stream>>>(
            xz, conv_w + (size_t)k * DD * 9, conv_b + k * DD, xc);
        // x_dbl projections (10 per position)
        k_xdbl<<<MM, 128, 0, stream>>>(xc, Wx + (size_t)k * 10 * DD, xd);
        // selective scan (direction k remap)
        k_scan<<<BB, 128, 0, stream>>>(
            xc, xd, Wdt + (size_t)k * DD * 8, dt_bias + k * DD,
            A_log + k * DD, Dp + k * DD, yb, k);
        // out-LN * silu(z-branch) -> f16
        k_outln<<<MM, 128, 0, stream>>>(yb, xz, onorm_w + k * DD, onorm_b + k * DD, g16);
        // @ Wout[k] into xm channel block k
        k_gemm_wmma<<<dim3(DD / 64, MM / 128), 128, 0, stream>>>(
            g16, DD, wout16 + (size_t)k * DD * DD, DD,
            xm + k * DD, CC, nullptr, DD, 0, LL, DD);
    }

    // 7) skip * SE * LN -> f16
    k_mixln<<<MM, 128, 0, stream>>>(xm, xsln, sebuf, skip_sc, norm_w, norm_b, xmn);

    // 8) final projection, transposed epilogue straight into (B, C, L) output
    k_gemm_wmma<<<dim3(CC / 64, MM / 128), 128, 0, stream>>>(
        xmn, CC, proj16, CC, out, 0, proj_b, CC, 1, LL, CC);
}
